// MultiHeadAttentionBSM3D_27891517620910
// MI455X (gfx1250) — matrix operand order repaired and checked by reading, not compiled, not run on hardware
//
#include <hip/hip_runtime.h>
#include <hip/hip_bf16.h>

// MultiHeadAttentionBSM3D for MI455X (gfx1250, wave32, WMMA).
// B=4, M=4, L=512, D=1024, H=16, DH=64. ROWS = B*M*L = 8192.

typedef _Float16 half_t;
typedef __attribute__((ext_vector_type(16))) _Float16 v16h;
typedef __attribute__((ext_vector_type(8)))  float    v8f;

// --------------------------------------------------------------------------
// CDNA5 async global->LDS copies (ASYNCcnt-tracked), with safe fallback.
// Builtin signature (from hipcc diagnostic): param0 = int4 addrspace(1)*,
// param1 = int4 addrspace(3)*, then imm offset, imm cpol.
// --------------------------------------------------------------------------
#if defined(__AMDGCN__) && \
    __has_builtin(__builtin_amdgcn_global_load_async_to_lds_b128) && \
    __has_builtin(__builtin_amdgcn_s_wait_asynccnt)
#define USE_ASYNC_LDS 1
#else
#define USE_ASYNC_LDS 0
#endif

#if USE_ASYNC_LDS
typedef int v4i_vec __attribute__((vector_size(16)));
typedef __attribute__((address_space(1))) v4i_vec as1_v4i;
typedef __attribute__((address_space(3))) v4i_vec as3_v4i;
#endif

// Copy 16 bytes (8 halfs) global -> LDS.
static __device__ __forceinline__ void copy_b128_to_lds(const half_t* g,
                                                        half_t* l) {
#if USE_ASYNC_LDS
  // Flat LDS addresses carry the DS offset in addr[31:0] (ISA 10.2), so the
  // integer round-trip recovers the addrspace(3) pointer value.
  __builtin_amdgcn_global_load_async_to_lds_b128(
      (as1_v4i*)(uintptr_t)g, (as3_v4i*)(uint32_t)(uintptr_t)l, 0, 0);
#else
  *(uint4*)l = *(const uint4*)g;
#endif
}

template <int N>
static __device__ __forceinline__ void wait_async() {
#if USE_ASYNC_LDS
  __builtin_amdgcn_s_wait_asynccnt((unsigned short)N);
#endif
}

static __device__ __forceinline__ v8f wmma16x16x32(v16h a, v16h b, v8f c) {
  // D = A(16x32 f16) x B(32x16 f16) + C(16x16 f32)
  return __builtin_amdgcn_wmma_f32_16x16x32_f16(
      /*neg_a=*/false, a, /*neg_b=*/false, b,
      /*c_mod=*/(short)0, c, /*reuse_a=*/false, /*reuse_b=*/false);
}

// ---------------------------------------------------------------------------
// Kernel 1: LayerNorm over last dim (1024) + cast to f16. One block per row.
// ---------------------------------------------------------------------------
__global__ __launch_bounds__(256) void ln_cast_kernel(
    const float* __restrict__ q, const float* __restrict__ g,
    const float* __restrict__ bb, half_t* __restrict__ out) {
  const int row = blockIdx.x;
  const int t = threadIdx.x;
  const float* x = q + (size_t)row * 1024;
  float s = 0.f, ss = 0.f;
  for (int i = t; i < 1024; i += 256) { float v = x[i]; s += v; ss += v * v; }
  __shared__ float rs[256], rq[256];
  rs[t] = s; rq[t] = ss; __syncthreads();
  for (int o = 128; o > 0; o >>= 1) {
    if (t < o) { rs[t] += rs[t + o]; rq[t] += rq[t + o]; }
    __syncthreads();
  }
  const float mean = rs[0] * (1.0f / 1024.0f);
  const float var = rq[0] * (1.0f / 1024.0f) - mean * mean;
  const float rstd = rsqrtf(var + 1e-6f);
  half_t* o = out + (size_t)row * 1024;
  for (int i = t; i < 1024; i += 256)
    o[i] = (half_t)(((x[i] - mean) * rstd) * g[i] + bb[i]);
}

// ---------------------------------------------------------------------------
// Kernel 2: f32 -> f16 cast (grid-stride)
// ---------------------------------------------------------------------------
__global__ __launch_bounds__(256) void cast_kernel(
    const float* __restrict__ in, half_t* __restrict__ out, int n) {
  int i = blockIdx.x * 256 + threadIdx.x;
  const int stride = gridDim.x * 256;
  for (; i < n; i += stride) out[i] = (half_t)in[i];
}

// ---------------------------------------------------------------------------
// Kernel 3: 128x64-tile GEMM, C[M,N] = A[M,K] @ Bw[K,N], f16 in, f32 accum.
// A tile streamed via async global->LDS B128 with double buffering; WMMA on
// tile i overlaps the async copy of tile i+1 (s_wait_asynccnt <= 2).
// MODE 0: fused per-64-col L2 normalize -> f16 (projection + normaliseE)
// MODE 2: f32 out + residual (final fc)
// Block = 256 threads = 8 waves as 4(M) x 2(N); each wave 32x32 (2x2 WMMA).
// ---------------------------------------------------------------------------
template <int MODE>
__global__ __launch_bounds__(256) void gemm128x64_kernel(
    const half_t* __restrict__ A, const half_t* __restrict__ Bw,
    half_t* __restrict__ Ch, float* __restrict__ Cf,
    const float* __restrict__ residual, int K, int N) {
  constexpr int BM = 128, BN = 64, BK = 32;
  constexpr int AST = BK + 8;  // 40 halfs = 80B rows: keeps 16B alignment
  __shared__ alignas(16) half_t As[2][BM][AST];
  __shared__ half_t Bs[2][BN][BK + 2];

  const int t = threadIdx.x;
  const int m0 = blockIdx.x * BM;
  const int n0 = blockIdx.y * BN;
  const int wave = t >> 5, lane = t & 31;
  const int wm = wave & 3, wn = wave >> 2;
  const int lhalf = lane >> 4, l16 = lane & 15;
  const int NT = K / BK;

  v8f acc[2][2];
  for (int i = 0; i < 2; i++)
    for (int j = 0; j < 2; j++)
      for (int e = 0; e < 8; e++) acc[i][j][e] = 0.f;

  // A tile: 128x32 halfs = 512 B128 chunks -> 2 async issues per lane.
  auto stageA = [&](int it, int buf) {
    const int kt = it * BK;
    for (int c = t; c < BM * (BK / 8); c += 256) {
      const int r = c >> 2, cc = (c & 3) << 3;
      copy_b128_to_lds(A + (size_t)(m0 + r) * K + kt + cc, &As[buf][r][cc]);
    }
  };
  // B tile 32x64, stored transposed (Bs[n][k]) so fragments are contiguous.
  auto stageB = [&](int it, int buf) {
    const int kt = it * BK;
    for (int c = t; c < BK * BN; c += 256) {
      const int k = c >> 6, n = c & 63;
      Bs[buf][n][k] = Bw[(size_t)(kt + k) * N + n0 + n];
    }
  };

  stageA(0, 0);
  stageB(0, 0);
  for (int it = 0; it < NT; ++it) {
    const int cur = it & 1;
    if (it + 1 < NT) {
      stageA(it + 1, cur ^ 1);
      stageB(it + 1, cur ^ 1);
      wait_async<2>();  // tile `it` complete; tile `it+1` may stay in flight
    } else {
      wait_async<0>();
    }
    __syncthreads();

    v16h af[2], bf[2];
#pragma unroll
    for (int tm = 0; tm < 2; tm++) {
      const int m = wm * 32 + tm * 16 + l16;
      const int kb = lhalf * 8;
#pragma unroll
      for (int j = 0; j < 8; j++) {
        af[tm][j] = As[cur][m][kb + j];
        af[tm][8 + j] = As[cur][m][16 + kb + j];
      }
    }
#pragma unroll
    for (int tn = 0; tn < 2; tn++) {
      const int n = wn * 32 + tn * 16 + l16;
      const int kb = lhalf * 8;
#pragma unroll
      for (int j = 0; j < 16; j++) bf[tn][j] = Bs[cur][n][kb + j + (j >= 8 ? 8 : 0)];
    }
#pragma unroll
    for (int tm = 0; tm < 2; tm++)
#pragma unroll
      for (int tn = 0; tn < 2; tn++)
        acc[tm][tn] = wmma16x16x32(af[tm], bf[tn], acc[tm][tn]);
    __syncthreads();
  }

  if constexpr (MODE == 0) {
    __shared__ float Cs[BM][BN + 1];
#pragma unroll
    for (int tm = 0; tm < 2; tm++)
      for (int tn = 0; tn < 2; tn++)
        for (int e = 0; e < 8; e++) {
          const int m = wm * 32 + tm * 16 + lhalf * 8 + e;
          const int n = wn * 32 + tn * 16 + l16;
          Cs[m][n] = acc[tm][tn][e];
        }
    __syncthreads();
    if (t < BM) {
      float ssq = 0.f;
      for (int c = 0; c < BN; c++) { const float v = Cs[t][c]; ssq += v * v; }
      const float inv = ssq > 0.f ? rsqrtf(ssq) : 0.f;
      half_t* dst = Ch + (size_t)(m0 + t) * N + n0;
      for (int c = 0; c < BN; c++) dst[c] = (half_t)(Cs[t][c] * inv);
    }
  } else {
#pragma unroll
    for (int tm = 0; tm < 2; tm++)
      for (int tn = 0; tn < 2; tn++)
        for (int e = 0; e < 8; e++) {
          const int gm = m0 + wm * 32 + tm * 16 + lhalf * 8 + e;
          const int gn = n0 + wn * 32 + tn * 16 + l16;
          if constexpr (MODE == 1)
            Ch[(size_t)gm * N + gn] = (half_t)acc[tm][tn][e];
          else
            Cf[(size_t)gm * N + gn] =
                acc[tm][tn][e] + residual[(size_t)gm * N + gn];
        }
  }
}

// ---------------------------------------------------------------------------
// Kernel 4: attention scores S = qh @ qih^T per (b,m,h), K=64, mask fused.
// Both tiles are row-contiguous copies -> async to LDS, double buffered
// (4 async issues per lane per tile). grid (4,4,256); 128x128 tile;
// 8 waves as 4(M) x 2(N), wave = 32x64 (2x4 WMMA).
// ---------------------------------------------------------------------------
__global__ __launch_bounds__(256) void score_kernel(
    const half_t* __restrict__ qh, const half_t* __restrict__ qih,
    const int* __restrict__ mask, float* __restrict__ attn) {
  constexpr int BT = 128, BK = 32;
  constexpr int AST = BK + 8;  // 80B rows
  const int z = blockIdx.z;
  const int bm = z >> 4, h = z & 15;
  const int q0 = blockIdx.x * BT, k0 = blockIdx.y * BT;
  const size_t rowbase = (size_t)bm * 512;
  const int col0 = h * 64;

  __shared__ alignas(16) half_t Aq[2][BT][AST];
  __shared__ alignas(16) half_t Bk[2][BT][AST];

  const int t = threadIdx.x;
  const int wave = t >> 5, lane = t & 31;
  const int wm = wave & 3, wn = wave >> 2;
  const int lhalf = lane >> 4, l16 = lane & 15;

  v8f acc[2][4];
  for (int i = 0; i < 2; i++)
    for (int j = 0; j < 4; j++)
      for (int e = 0; e < 8; e++) acc[i][j][e] = 0.f;

  auto stage = [&](int it, int buf) {
    const int kt = it * BK;
    for (int c = t; c < BT * (BK / 8); c += 256) {
      const int r = c >> 2, cc = (c & 3) << 3;
      copy_b128_to_lds(qh + (rowbase + q0 + r) * 1024 + col0 + kt + cc,
                       &Aq[buf][r][cc]);
      copy_b128_to_lds(qih + (rowbase + k0 + r) * 1024 + col0 + kt + cc,
                       &Bk[buf][r][cc]);
    }
  };

  stage(0, 0);
  for (int it = 0; it < 2; ++it) {  // K = 64 = 2 * BK
    const int cur = it & 1;
    if (it + 1 < 2) {
      stage(it + 1, cur ^ 1);
      wait_async<4>();
    } else {
      wait_async<0>();
    }
    __syncthreads();

    v16h af[2], bf[4];
#pragma unroll
    for (int tm = 0; tm < 2; tm++) {
      const int m = wm * 32 + tm * 16 + l16;
      const int kb = lhalf * 8;
#pragma unroll
      for (int j = 0; j < 8; j++) {
        af[tm][j] = Aq[cur][m][kb + j];
        af[tm][8 + j] = Aq[cur][m][16 + kb + j];
      }
    }
#pragma unroll
    for (int tn = 0; tn < 4; tn++) {
      const int n = wn * 64 + tn * 16 + l16;
      const int kb = lhalf * 8;
#pragma unroll
      for (int j = 0; j < 16; j++) bf[tn][j] = Bk[cur][n][kb + j + (j >= 8 ? 8 : 0)];
    }
#pragma unroll
    for (int tm = 0; tm < 2; tm++)
#pragma unroll
      for (int tn = 0; tn < 4; tn++)
        acc[tm][tn] = wmma16x16x32(af[tm], bf[tn], acc[tm][tn]);
    __syncthreads();
  }

  const size_t abase = (size_t)z * 512 * 512;
  const size_t mbase = (size_t)bm * 512 * 512;
#pragma unroll
  for (int tm = 0; tm < 2; tm++)
    for (int tn = 0; tn < 4; tn++)
      for (int e = 0; e < 8; e++) {
        const int qq = q0 + wm * 32 + tm * 16 + lhalf * 8 + e;
        const int kk = k0 + wn * 64 + tn * 16 + l16;
        const size_t idx = (size_t)qq * 512 + kk;
        attn[abase + idx] = (mask[mbase + idx] == 0) ? -1e9f : acc[tm][tn][e];
      }
}

// ---------------------------------------------------------------------------
// Kernel 5: joint softmax over (m, k): 2048 values per (b,h,q), in-place.
// One block per (b,h,q); values held in registers -> one read + one write.
// ---------------------------------------------------------------------------
__global__ __launch_bounds__(256) void softmax_kernel(float* __restrict__ attn) {
  const int bid = blockIdx.x;
  const int b = bid >> 13;          // / (16*512)
  const int h = (bid >> 9) & 15;    // / 512 % 16
  const int qq = bid & 511;
  const int t = threadIdx.x;

  size_t bases[4];
#pragma unroll
  for (int mm = 0; mm < 4; mm++)
    bases[mm] = (((size_t)((b * 4 + mm) * 16 + h) * 512 + qq) * 512);

  float v[8];
  int idx = 0;
#pragma unroll
  for (int mm = 0; mm < 4; mm++) {
    v[idx++] = attn[bases[mm] + t];
    v[idx++] = attn[bases[mm] + t + 256];
  }
  float mx = v[0];
#pragma unroll
  for (int j = 1; j < 8; j++) mx = fmaxf(mx, v[j]);

  __shared__ float red[256];
  red[t] = mx; __syncthreads();
  for (int o = 128; o > 0; o >>= 1) {
    if (t < o) red[t] = fmaxf(red[t], red[t + o]);
    __syncthreads();
  }
  mx = red[0]; __syncthreads();

  float s = 0.f;
#pragma unroll
  for (int j = 0; j < 8; j++) { v[j] = __expf(v[j] - mx); s += v[j]; }
  red[t] = s; __syncthreads();
  for (int o = 128; o > 0; o >>= 1) {
    if (t < o) red[t] += red[t + o];
    __syncthreads();
  }
  const float inv = 1.f / red[0];

  idx = 0;
#pragma unroll
  for (int mm = 0; mm < 4; mm++) {
    attn[bases[mm] + t] = v[idx++] * inv;
    attn[bases[mm] + t + 256] = v[idx++] * inv;
  }
}

// ---------------------------------------------------------------------------
// Kernel 6: ctx = attn @ qih per (b,m,h): (512x512)x(512x64), f32 attn
// converted to f16 while staging to LDS (conversion precludes async copy).
// grid (4,1,256).
// ---------------------------------------------------------------------------
__global__ __launch_bounds__(256) void ctx_kernel(
    const float* __restrict__ attn, const half_t* __restrict__ qih,
    half_t* __restrict__ ctx) {
  constexpr int BM = 128, BN = 64, BK = 32;
  const int z = blockIdx.z;
  const int bm = z >> 4, h = z & 15;
  const int q0 = blockIdx.x * BM;
  const size_t abase = (size_t)z * 512 * 512;
  const size_t rowbase = (size_t)bm * 512;
  const int col0 = h * 64;

  __shared__ half_t As[BM][BK + 2];
  __shared__ half_t Bs[BN][BK + 2];

  const int t = threadIdx.x;
  const int wave = t >> 5, lane = t & 31;
  const int wm = wave & 3, wn = wave >> 2;
  const int lhalf = lane >> 4, l16 = lane & 15;

  v8f acc[2][2];
  for (int i = 0; i < 2; i++)
    for (int j = 0; j < 2; j++)
      for (int e = 0; e < 8; e++) acc[i][j][e] = 0.f;

  for (int kt = 0; kt < 512; kt += BK) {
    for (int c = t; c < BM * BK; c += 256) {
      const int r = c >> 5, cc = c & 31;
      As[r][cc] = (half_t)attn[abase + (size_t)(q0 + r) * 512 + kt + cc];
    }
    for (int c = t; c < BK * BN; c += 256) {
      const int k = c >> 6, n = c & 63;
      Bs[n][k] = qih[(rowbase + kt + k) * 1024 + col0 + n];
    }
    __syncthreads();

    v16h af[2], bf[2];
#pragma unroll
    for (int tm = 0; tm < 2; tm++) {
      const int m = wm * 32 + tm * 16 + l16;
      const int kb = lhalf * 8;
#pragma unroll
      for (int j = 0; j < 8; j++) {
        af[tm][j] = As[m][kb + j];
        af[tm][8 + j] = As[m][16 + kb + j];
      }
    }
#pragma unroll
    for (int tn = 0; tn < 2; tn++) {
      const int n = wn * 32 + tn * 16 + l16;
      const int kb = lhalf * 8;
#pragma unroll
      for (int j = 0; j < 16; j++) bf[tn][j] = Bs[n][kb + j + (j >= 8 ? 8 : 0)];
    }
#pragma unroll
    for (int tm = 0; tm < 2; tm++)
#pragma unroll
      for (int tn = 0; tn < 2; tn++)
        acc[tm][tn] = wmma16x16x32(af[tm], bf[tn], acc[tm][tn]);
    __syncthreads();
  }

#pragma unroll
  for (int tm = 0; tm < 2; tm++)
    for (int tn = 0; tn < 2; tn++)
      for (int e = 0; e < 8; e++) {
        const int gm = q0 + wm * 32 + tm * 16 + lhalf * 8 + e;
        const int gn = wn * 32 + tn * 16 + l16;
        ctx[(rowbase + gm) * 1024 + col0 + gn] = (half_t)acc[tm][tn][e];
      }
}

// ---------------------------------------------------------------------------
extern "C" void kernel_launch(void* const* d_in, const int* in_sizes, int n_in,
                              void* d_out, int out_size, void* d_ws,
                              size_t ws_size, hipStream_t stream) {
  (void)in_sizes; (void)n_in; (void)out_size; (void)ws_size;

  const float* q    = (const float*)d_in[0];
  // d_in[1] = k, unused by the reference forward
  const float* qi   = (const float*)d_in[2];
  const int*   mask = (const int*)d_in[3];
  const float* wqs  = (const float*)d_in[4];
  const float* wqis = (const float*)d_in[5];
  const float* wfc  = (const float*)d_in[6];
  const float* lng  = (const float*)d_in[7];
  const float* lnb  = (const float*)d_in[8];

  float* outp = (float*)d_out;                      // [8192,1024]
  float* attn = outp + (size_t)8192 * 1024;         // [256,512,512]

  char* w = (char*)d_ws;
  auto alloc = [&](size_t bytes) { void* p = (void*)w; w += bytes; return p; };
  half_t* qn16   = (half_t*)alloc((size_t)8192 * 1024 * 2);
  half_t* qi16   = (half_t*)alloc((size_t)8192 * 1024 * 2);
  half_t* qh16   = (half_t*)alloc((size_t)8192 * 1024 * 2);
  half_t* qih16  = (half_t*)alloc((size_t)8192 * 1024 * 2);
  half_t* ctx16  = (half_t*)alloc((size_t)8192 * 1024 * 2);
  half_t* wqs16  = (half_t*)alloc((size_t)1024 * 1024 * 2);
  half_t* wqis16 = (half_t*)alloc((size_t)1024 * 1024 * 2);
  half_t* wfc16  = (half_t*)alloc((size_t)1024 * 1024 * 2);

  // 1) LayerNorm(q) -> f16; casts for qi and weights
  ln_cast_kernel<<<8192, 256, 0, stream>>>(q, lng, lnb, qn16);
  cast_kernel<<<2048, 256, 0, stream>>>(qi, qi16, 8192 * 1024);
  cast_kernel<<<1024, 256, 0, stream>>>(wqs, wqs16, 1024 * 1024);
  cast_kernel<<<1024, 256, 0, stream>>>(wqis, wqis16, 1024 * 1024);
  cast_kernel<<<1024, 256, 0, stream>>>(wfc, wfc16, 1024 * 1024);

  // 2) projections with fused per-head L2 normalize (MODE 0)
  dim3 g1(64, 16);
  gemm128x64_kernel<0><<<g1, 256, 0, stream>>>(qn16, wqs16, qh16, nullptr,
                                               nullptr, 1024, 1024);
  gemm128x64_kernel<0><<<g1, 256, 0, stream>>>(qi16, wqis16, qih16, nullptr,
                                               nullptr, 1024, 1024);

  // 3) cosine scores + mask -> attn logits (written into d_out attn region)
  dim3 gs(4, 4, 256);
  score_kernel<<<gs, 256, 0, stream>>>(qh16, qih16, mask, attn);

  // 4) joint softmax over (m, k), in place
  softmax_kernel<<<32768, 256, 0, stream>>>(attn);

  // 5) ctx = attn @ qih
  dim3 gc(4, 1, 256);
  ctx_kernel<<<gc, 256, 0, stream>>>(attn, qih16, ctx16);

  // 6) out = ctx @ w_fc + residual(q)  (MODE 2)
  gemm128x64_kernel<2><<<g1, 256, 0, stream>>>(ctx16, wfc16, nullptr, outp, q,
                                               1024, 1024);
}
